// BahdanauAttention_10651518894570
// MI455X (gfx1250) — matrix operand order repaired and checked by reading, not compiled, not run on hardware
//
#include <hip/hip_runtime.h>
#include <hip/hip_bf16.h>

// ---------------- problem constants ----------------
#define BB 32
#define SS 2048
#define HH 1024
#define ROWS 128                 // s-rows per workgroup in the GEMM kernel
#define LDS_STRIDE (HH + 8)      // padded bf16 row stride to spread LDS banks
#define LDS_BYTES (ROWS * LDS_STRIDE * 2)   // 264192 B < 320 KB/WGP
#define NSPLIT 8                 // S-splits for the context GEMV

typedef __attribute__((ext_vector_type(16))) __bf16 v16bf;
typedef __attribute__((ext_vector_type(8)))  float  v8f;

struct alignas(32) BF16x16 { unsigned int u[8]; };

static __device__ __forceinline__ unsigned short f32_to_bf16(float f) {
    unsigned int u = __builtin_bit_cast(unsigned int, f);
    u += 0x7FFFu + ((u >> 16) & 1u);     // round-to-nearest-even
    return (unsigned short)(u >> 16);
}

// branchless tanh: clamp then (e^2x - 1) * rcp(e^2x + 1); v_exp + v_rcp
static __device__ __forceinline__ float fast_tanh(float x) {
    x = fminf(fmaxf(x, -9.f), 9.f);
    float e = __expf(2.f * x);
    return (e - 1.f) * __builtin_amdgcn_rcpf(e + 1.f);
}

// A 16x32 bf16 fragment from LDS:
// lanes 0-15 -> K pairs {0,1}..{6,7}; lanes 16-31 -> +8; vgprs 4-7 -> +16
static __device__ __forceinline__ void load_a(
        BF16x16& d, const unsigned short* __restrict__ arow, int kb, int lhalf) {
    #pragma unroll
    for (int i = 0; i < 8; ++i) {
        int kk = kb + ((i & 4) << 2) + (lhalf << 3) + ((i & 3) << 1);
        d.u[i] = *(const unsigned int*)&arow[kk];
    }
}

// B 32x16 bf16 fragment from global (L2-hot bf16 Wk):
// lane -> N column, lanes 0-15 -> K pairs {0,1}..{6,7}; lanes 16-31 -> +8; vgprs 4-7 -> +16
static __device__ __forceinline__ void load_b(
        BF16x16& d, const unsigned int* __restrict__ brow, int kb, int lhalf) {
    #pragma unroll
    for (int i = 0; i < 8; ++i) {
        int kk = kb + ((i & 4) << 2) + (lhalf << 3) + ((i & 3) << 1);
        d.u[i] = brow[kk >> 1];
    }
}

static __device__ __forceinline__ v8f wmma_bf16(const BF16x16& a, const BF16x16& b, v8f c) {
    return __builtin_amdgcn_wmma_f32_16x16x32_bf16(
        false, __builtin_bit_cast(v16bf, a),
        false, __builtin_bit_cast(v16bf, b),
        (short)0, c, false, false);
}

// ---------------- K0: Wk fp32 -> bf16 ----------------
__global__ __launch_bounds__(256) void convert_wk_kernel(
        const float* __restrict__ Wk, unsigned short* __restrict__ out) {
    int i = (blockIdx.x * 256 + threadIdx.x) * 4;
    #pragma unroll
    for (int j = 0; j < 4; ++j) out[i + j] = f32_to_bf16(Wk[i + j]);
}

// ---------------- K1: proj_query[b][h] = dot(query[b], Wq[h]) ----------------
__global__ __launch_bounds__(256) void proj_query_kernel(
        const float* __restrict__ query, const float* __restrict__ Wq,
        float* __restrict__ pq) {
    int gw   = (blockIdx.x * 256 + threadIdx.x) >> 5;   // global wave id = b*H+h
    int lane = threadIdx.x & 31;
    int b = gw >> 10, h = gw & (HH - 1);
    const float* q = query + b * HH;
    const float* w = Wq + h * HH;
    float s = 0.f;
    for (int k = lane; k < HH; k += 32) s += q[k] * w[k];
    #pragma unroll
    for (int m = 16; m; m >>= 1) s += __shfl_xor(s, m, 32);
    if (lane == 0) pq[gw] = s;
}

// ---------------- K2: fused keys@Wk^T -> tanh -> dot(We) scores ----------------
// 8 waves / 256 threads; wave owns 16 s-rows; 2 N-tiles per pass (A fragment
// reused -> half LDS traffic); double-buffered software pipeline with
// SPLIT even/odd accumulators so every acc register is written exactly once
// per loop iteration (no regalloc copies, no WMMA->WMMA chaining pressure).
__global__ __launch_bounds__(256) void score_gemm_kernel(
        const float* __restrict__ keys, const unsigned short* __restrict__ wkb,
        const float* __restrict__ pq, const float* __restrict__ We,
        float* __restrict__ escore) {
    extern __shared__ unsigned short lds[];   // ROWS x LDS_STRIDE bf16
    const int b  = blockIdx.y;
    const int s0 = blockIdx.x * ROWS;
    const int tid = threadIdx.x;

    // stage 128x1024 keys tile into LDS as bf16 (HBM read exactly once)
    // float4 loads, packed 4xbf16 (b64) LDS stores
    const float* kp = keys + ((size_t)b * SS + s0) * HH;
    const float4* kp4 = (const float4*)kp;
    for (int i = tid; i < ROWS * HH / 4; i += 256) {
        float4 v = kp4[i];
        int idx = i * 4;
        int row = idx >> 10, col = idx & (HH - 1);
        unsigned int lo = (unsigned int)f32_to_bf16(v.x) |
                          ((unsigned int)f32_to_bf16(v.y) << 16);
        unsigned int hi = (unsigned int)f32_to_bf16(v.z) |
                          ((unsigned int)f32_to_bf16(v.w) << 16);
        unsigned long long p = ((unsigned long long)hi << 32) | lo;
        *(unsigned long long*)&lds[row * LDS_STRIDE + col] = p;
    }
    __syncthreads();

    const int wave  = tid >> 5;
    const int lane  = tid & 31;
    const int m0    = wave * 16;       // this wave's 16 rows
    const int lhalf = lane >> 4;       // 0: lanes 0-15, 1: lanes 16-31
    const int l16   = lane & 15;

    const unsigned int* wkb32 = (const unsigned int*)wkb;  // 512 dwords / Wk row
    const unsigned short* arow = &lds[(m0 + l16) * LDS_STRIDE];
    const float* pqb = pq + b * HH;

    float sacc[8] = {0.f, 0.f, 0.f, 0.f, 0.f, 0.f, 0.f, 0.f};

    for (int nb = 0; nb < HH / 16; nb += 2) {
        const int n0 = nb * 16 + l16;
        const unsigned int* brow0 = wkb32 + n0 * (HH / 2);
        const unsigned int* brow1 = wkb32 + (n0 + 16) * (HH / 2);

        // issue epilogue scalars early; VMEM loads retire in order,
        // so these are complete long before the epilogue reads them
        const float pq0 = pqb[n0],      we0 = We[n0];
        const float pq1 = pqb[n0 + 16], we1 = We[n0 + 16];

        // even/odd phase accumulators: each written once per iteration
        v8f acc00 = {}, acc10 = {};     // even phase (A0/B*0)
        v8f acc01 = {}, acc11 = {};     // odd  phase (A1/B*1)

        BF16x16 A0, B00, B10, A1, B01, B11;
        load_a(A0, arow, 0, lhalf);
        load_b(B00, brow0, 0, lhalf);
        load_b(B10, brow1, 0, lhalf);

        for (int kb = 0; kb < HH - 64; kb += 64) {
            load_a(A1, arow, kb + 32, lhalf);
            load_b(B01, brow0, kb + 32, lhalf);
            load_b(B11, brow1, kb + 32, lhalf);
            acc00 = wmma_bf16(A0, B00, acc00);   // waits only on phase-old loads
            acc10 = wmma_bf16(A0, B10, acc10);

            load_a(A0, arow, kb + 64, lhalf);
            load_b(B00, brow0, kb + 64, lhalf);
            load_b(B10, brow1, kb + 64, lhalf);
            acc01 = wmma_bf16(A1, B01, acc01);
            acc11 = wmma_bf16(A1, B11, acc11);
        }
        // peeled tail: A0 set holds chunk HH-64
        load_a(A1, arow, HH - 32, lhalf);
        load_b(B01, brow0, HH - 32, lhalf);
        load_b(B11, brow1, HH - 32, lhalf);
        acc00 = wmma_bf16(A0, B00, acc00);
        acc10 = wmma_bf16(A0, B10, acc10);
        acc01 = wmma_bf16(A1, B01, acc01);
        acc11 = wmma_bf16(A1, B11, acc11);

        // fused epilogue: tanh(pk + pq) * We, accumulated per row (branchless)
        #pragma unroll
        for (int r = 0; r < 8; ++r) {
            float p0 = acc00[r] + acc01[r];
            float p1 = acc10[r] + acc11[r];
            sacc[r] += fast_tanh(p0 + pq0) * we0 + fast_tanh(p1 + pq1) * we1;
        }
    }

    // C/D layout: vgpr r, lanes 0-15 -> row r, lanes 16-31 -> row r+8.
    // Reduce over N (16 lanes per half-wave), deterministic tree.
    #pragma unroll
    for (int r = 0; r < 8; ++r) {
        float v = sacc[r];
        v += __shfl_xor(v, 8, 16);
        v += __shfl_xor(v, 4, 16);
        v += __shfl_xor(v, 2, 16);
        v += __shfl_xor(v, 1, 16);
        if (l16 == 0)
            escore[b * SS + s0 + m0 + r + lhalf * 8] = v;
    }
}

// ---------------- K3: masked softmax over S per batch ----------------
__global__ __launch_bounds__(256) void softmax_kernel(
        const float* __restrict__ escore, const int* __restrict__ mask,
        float* __restrict__ alphas) {
    __shared__ float red[256];
    const int b = blockIdx.x, tid = threadIdx.x;
    const float NEG_INF = -__builtin_huge_valf();
    float vals[SS / 256];
    float mx = NEG_INF;
    #pragma unroll
    for (int i = 0; i < SS / 256; ++i) {
        int s = tid + i * 256;
        float v = (mask[b * SS + s] == 0) ? NEG_INF : escore[b * SS + s];
        vals[i] = v;
        mx = fmaxf(mx, v);
    }
    red[tid] = mx; __syncthreads();
    for (int off = 128; off > 0; off >>= 1) {
        if (tid < off) red[tid] = fmaxf(red[tid], red[tid + off]);
        __syncthreads();
    }
    mx = red[0]; __syncthreads();
    float sum = 0.f;
    #pragma unroll
    for (int i = 0; i < SS / 256; ++i) {
        float ex = (vals[i] == NEG_INF) ? 0.f : expf(vals[i] - mx);
        vals[i] = ex;
        sum += ex;
    }
    red[tid] = sum; __syncthreads();
    for (int off = 128; off > 0; off >>= 1) {
        if (tid < off) red[tid] += red[tid + off];
        __syncthreads();
    }
    float inv = 1.f / red[0];
    #pragma unroll
    for (int i = 0; i < SS / 256; ++i)
        alphas[b * SS + tid + i * 256] = vals[i] * inv;
}

// ---------------- K4: context partials (split-S weighted sum of values) ----------------
__global__ __launch_bounds__(256) void context_partial_kernel(
        const float* __restrict__ alphas, const float* __restrict__ values,
        float* __restrict__ part) {
    __shared__ float al[SS / NSPLIT];
    const int b = blockIdx.z, sc = blockIdx.y, tid = threadIdx.x;
    const int sbase = sc * (SS / NSPLIT);
    al[tid] = alphas[b * SS + sbase + tid];
    __syncthreads();
    const int h = blockIdx.x * 256 + tid;
    const float* vp = values + ((size_t)b * SS + sbase) * HH + h;
    float sum = 0.f;
    #pragma unroll 4
    for (int j = 0; j < SS / NSPLIT; ++j)
        sum += al[j] * vp[(size_t)j * HH];
    part[(b * NSPLIT + sc) * HH + h] = sum;
}

// ---------------- K5: reduce partials -> context ----------------
__global__ __launch_bounds__(256) void context_reduce_kernel(
        const float* __restrict__ part, float* __restrict__ ctx) {
    int idx = blockIdx.x * 256 + threadIdx.x;       // 0 .. B*H-1
    int b = idx >> 10, h = idx & (HH - 1);
    float s = 0.f;
    #pragma unroll
    for (int c = 0; c < NSPLIT; ++c) s += part[(b * NSPLIT + c) * HH + h];
    ctx[idx] = s;
}

// ---------------- launch ----------------
extern "C" void kernel_launch(void* const* d_in, const int* in_sizes, int n_in,
                              void* d_out, int out_size, void* d_ws, size_t ws_size,
                              hipStream_t stream) {
    const float* query  = (const float*)d_in[0];   // (B,1,H)
    const int*   mask   = (const int*)  d_in[1];   // (B,1,S)
    const float* values = (const float*)d_in[2];   // (B,S,H)
    const float* keys   = (const float*)d_in[3];   // (B,S,H)
    const float* Wk     = (const float*)d_in[4];   // (H,H)
    const float* Wq     = (const float*)d_in[5];   // (H,H)
    const float* We     = (const float*)d_in[6];   // (1,H)

    char* ws = (char*)d_ws;
    unsigned short* wkbf = (unsigned short*)ws;                      // 2 MB
    float* pq    = (float*)(ws + (size_t)HH * HH * 2);               // 128 KB
    float* esc   = (float*)(ws + (size_t)HH * HH * 2 + BB * HH * 4); // 256 KB
    float* part  = (float*)(ws + (size_t)HH * HH * 2 + BB * HH * 4 + BB * SS * 4);

    float* ctx    = (float*)d_out;            // (B,1,H) first
    float* alphas = (float*)d_out + BB * HH;  // (B,1,S) second

    hipFuncSetAttribute(reinterpret_cast<const void*>(score_gemm_kernel),
                        hipFuncAttributeMaxDynamicSharedMemorySize, LDS_BYTES);

    convert_wk_kernel<<<(HH * HH) / (256 * 4), 256, 0, stream>>>(Wk, wkbf);
    proj_query_kernel<<<(BB * HH * 32) / 256, 256, 0, stream>>>(query, Wq, pq);
    score_gemm_kernel<<<dim3(SS / ROWS, BB), 256, LDS_BYTES, stream>>>(
        keys, wkbf, pq, We, esc);
    softmax_kernel<<<BB, 256, 0, stream>>>(esc, mask, alphas);
    context_partial_kernel<<<dim3(HH / 256, NSPLIT, BB), 256, 0, stream>>>(
        alphas, values, part);
    context_reduce_kernel<<<(BB * HH) / 256, 256, 0, stream>>>(part, ctx);
}
